// LSTMAddonsDecoder_56143812493671
// MI455X (gfx1250) — matrix operand order repaired and checked by reading, not compiled, not run on hardware
//
#include <hip/hip_runtime.h>
#include <hip/hip_bf16.h>

typedef __bf16 bf16_t;
typedef __attribute__((ext_vector_type(16))) __bf16 v16bf;
typedef __attribute__((ext_vector_type(8)))  __bf16 v8bf;
typedef __attribute__((ext_vector_type(8)))  float  v8f;

#define B_    64
#define T_    512
#define DIN   256
#define H_    1024
#define DOUT  256
#define KTOT  1280          // DIN + H
#define NKT   40            // KTOT/32  k-tiles in recurrent GEMM
#define XKT   8             // DIN/32   k-tiles from x
#define HKT   32            // H/32     k-tiles from h
#define NGT   64            // H/16     n-tiles per gate
#define HS_SLOT (4*HKT*512) // 65536 bf16 elements per h snapshot
#define NWG   16            // workgroups in the persistent step kernel

#define WMMA_BF16(a, b, c) \
  __builtin_amdgcn_wmma_f32_16x16x32_bf16(false, (a), false, (b), (short)0, (c), false, false)

__device__ __forceinline__ float sigm(float x)      { return 1.0f / (1.0f + __expf(-x)); }
__device__ __forceinline__ float tanh_fast(float x) { return 2.0f * sigm(2.0f * x) - 1.0f; }

// ---------------- prep: [Wx;Wh] (1280x4096) -> B-fragment-ready bf16 ----------------
// B layout (32x16 tile): lane l holds col n=l&15, j<8 -> K = (l>>4)*8 + j ; j>=8 -> K = 16 + (l>>4)*8 + (j-8)
__global__ void prep_wb(const float* __restrict__ Wx, const float* __restrict__ Wh,
                        bf16_t* __restrict__ WB, int total) {
  int idx = blockIdx.x * blockDim.x + threadIdx.x;
  if (idx >= total) return;
  int within = idx & 511, fi = idx >> 9;
  int lane = within >> 4, j = within & 15;
  int nt = fi & 255, kt = fi >> 8;
  int k = kt * 32 + ((j < 8) ? ((lane >> 4) * 8 + j) : (16 + (lane >> 4) * 8 + (j - 8)));
  int n = nt * 16 + (lane & 15);
  float v = (k < DIN) ? Wx[(size_t)k * 4096 + n] : Wh[(size_t)(k - DIN) * 4096 + n];
  WB[idx] = (bf16_t)v;
}

// ---------------- prep: Wo (1024x256) -> B-fragment-ready ----------------
__global__ void prep_wob(const float* __restrict__ Wo, bf16_t* __restrict__ WoB, int total) {
  int idx = blockIdx.x * blockDim.x + threadIdx.x;
  if (idx >= total) return;
  int within = idx & 511, fi = idx >> 9;
  int lane = within >> 4, j = within & 15;
  int nt = fi & 15, kt = fi >> 4;
  int k = kt * 32 + ((j < 8) ? ((lane >> 4) * 8 + j) : (16 + (lane >> 4) * 8 + (j - 8)));
  int o = nt * 16 + (lane & 15);
  WoB[idx] = (bf16_t)Wo[(size_t)k * 256 + o];
}

// ---------------- prep: x -> A-fragment-ready bf16, per timestep ----------------
// A layout (16x32 tile): lane l (grp=l>>4) holds row m=l&15;
//   j<8 -> K = grp*8 + j ;  j>=8 -> K = 16 + grp*8 + (j-8)
__global__ void prep_xa(const float* __restrict__ x, bf16_t* __restrict__ xA, int total) {
  int idx = blockIdx.x * blockDim.x + threadIdx.x;
  if (idx >= total) return;
  int within = idx & 511, fi = idx >> 9;
  int lane = within >> 4, j = within & 15;
  int kt = fi & 7, mt = (fi >> 3) & 3, t = fi >> 5;
  int grp = lane >> 4;
  int kk = kt * 32 + ((j < 8) ? (grp * 8 + j) : (16 + grp * 8 + (j - 8)));
  int btc = mt * 16 + (lane & 15);
  xA[idx] = (bf16_t)x[((size_t)btc * T_ + t) * DIN + kk];
}

// ---------------- init h0 fragments and inter-step barrier slots ----------------
__global__ void init_state(bf16_t* __restrict__ hs0, int* __restrict__ bar, int n) {
  int idx = blockIdx.x * blockDim.x + threadIdx.x;
  if (idx >= n) return;
  hs0[idx] = (bf16_t)0.0f;
  if (idx < T_) bar[idx] = 0;
}

// ---------------- persistent LSTM: all 512 timesteps in one launch ----------------
// 16 workgroups; wg owns h columns [wg*64, wg*64+64).
// 8 waves: wave w -> (mt-pair p=w&1 -> mt {2p,2p+1}, column slice i=w>>1).
// Per k-tile: B fragments reused across 2 m-tiles, A fragments across 4 gates.
// h-loop uses an explicit 1-deep double buffer (unroll 2 to avoid rotation moves).
// Cell state lives in registers across all timesteps.
// Steps are separated by a device-scope release/acquire spin barrier on bar[t].
__global__ __launch_bounds__(256) void lstm_persistent(const bf16_t* __restrict__ xA,
                                                       const bf16_t* __restrict__ WB,
                                                       const float* __restrict__ bias,
                                                       bf16_t* __restrict__ hs,
                                                       int* __restrict__ bar) {
  __shared__ bf16_t hlds[64 * 64];  // h_new tile of this wg, bf16, [m][local col]
  int tid = threadIdx.x, wg = blockIdx.x;
  int w = tid >> 5, lane = tid & 31;
  int p = w & 1, i = w >> 1;  // p: m-tile pair, i: 16-col slice (0..3)
  int lane16 = lane * 16;
  int hgrp = lane >> 4;

  const size_t BK = (size_t)256 * 512;   // B stride per k-tile
  const size_t BG = (size_t)NGT * 512;   // B offset per gate
  const size_t AX = (size_t)XKT * 512;   // A stride between m-tiles in xA
  const size_t AH = (size_t)HKT * 512;   // A stride between m-tiles in hs

  // hoisted per-wave constants
  int colg = wg * 64 + i * 16 + (lane & 15);
  float bi  = bias[colg];
  float bf_ = bias[H_ + colg];
  float bg  = bias[2 * H_ + colg];
  float bo_ = bias[3 * H_ + colg];

  // cell state in registers: creg[m-tile][v]
  float creg[2][8];
#pragma unroll
  for (int m2 = 0; m2 < 2; ++m2)
#pragma unroll
    for (int v = 0; v < 8; ++v) creg[m2][v] = 0.0f;

  for (int t = 0; t < T_; ++t) {
    const bf16_t* ax = xA + (((size_t)t * 4 + 2 * p) * XKT) * 512 + lane16;
    const bf16_t* ah = hs + (size_t)t * HS_SLOT + ((size_t)(2 * p) * HKT) * 512 + lane16;
    const bf16_t* bp = WB + ((size_t)(wg * 4 + i)) * 512 + lane16;

    v8f acc[2][4] = {};  // [m-tile][gate i,f,g,o]

    // ---- x contribution: k-tiles 0..7, simple body (low register pressure) ----
#pragma unroll 2
    for (int kt = 0; kt < XKT; ++kt) {
      v16bf a0 = *(const v16bf*)ax;
      v16bf a1 = *(const v16bf*)(ax + AX);
      v16bf bb[4];
#pragma unroll
      for (int g = 0; g < 4; ++g) bb[g] = *(const v16bf*)(bp + g * BG);
#pragma unroll
      for (int g = 0; g < 4; ++g) {
        acc[0][g] = WMMA_BF16(a0, bb[g], acc[0][g]);
        acc[1][g] = WMMA_BF16(a1, bb[g], acc[1][g]);
      }
      ax += 512; bp += BK;
    }

    // ---- h contribution: k-tiles 8..39, explicit 1-deep double buffer ----
    v16bf a_c[2], b_c[4];
    a_c[0] = *(const v16bf*)ah;
    a_c[1] = *(const v16bf*)(ah + AH);
#pragma unroll
    for (int g = 0; g < 4; ++g) b_c[g] = *(const v16bf*)(bp + g * BG);
#pragma unroll 2
    for (int kt = 0; kt < HKT - 1; ++kt) {
      ah += 512; bp += BK;
      v16bf a_n[2], b_n[4];
      a_n[0] = *(const v16bf*)ah;
      a_n[1] = *(const v16bf*)(ah + AH);
#pragma unroll
      for (int g = 0; g < 4; ++g) b_n[g] = *(const v16bf*)(bp + g * BG);
#pragma unroll
      for (int g = 0; g < 4; ++g) {
        acc[0][g] = WMMA_BF16(a_c[0], b_c[g], acc[0][g]);
        acc[1][g] = WMMA_BF16(a_c[1], b_c[g], acc[1][g]);
      }
      a_c[0] = a_n[0]; a_c[1] = a_n[1];
#pragma unroll
      for (int g = 0; g < 4; ++g) b_c[g] = b_n[g];
    }
#pragma unroll
    for (int g = 0; g < 4; ++g) {
      acc[0][g] = WMMA_BF16(a_c[0], b_c[g], acc[0][g]);
      acc[1][g] = WMMA_BF16(a_c[1], b_c[g], acc[1][g]);
    }

    // gate math (D layout: lane holds col n=l&15, rows m = v + 8*(l>>4)); c in registers
#pragma unroll
    for (int m2 = 0; m2 < 2; ++m2) {
      int mt = 2 * p + m2;
#pragma unroll
      for (int v = 0; v < 8; ++v) {
        int m = mt * 16 + v + 8 * hgrp;
        float ig = sigm(acc[m2][0][v] + bi);
        float fg = sigm(acc[m2][1][v] + bf_);
        float gg = tanh_fast(acc[m2][2][v] + bg);
        float og = sigm(acc[m2][3][v] + bo_);
        float cn = fg * creg[m2][v] + ig * gg;
        creg[m2][v] = cn;
        float hv = og * tanh_fast(cn);
        hlds[m * 64 + i * 16 + (lane & 15)] = (bf16_t)hv;
      }
    }
    __syncthreads();

    // repack h_new (64 x 64 bf16) -> A-fragment layout for k-tiles {wg*2, wg*2+1}
    // 8 fragments (4 m-tiles x 2 k-tiles): exactly one per wave.
    {
      int mt2 = w & 3, ktl = w >> 2;
      int ml = lane & 15, grp = lane >> 4;
      const bf16_t* src = &hlds[(mt2 * 16 + ml) * 64 + ktl * 32 + grp * 8];
      v8bf lo = *(const v8bf*)src;         // j 0..7  : K = grp*8 + 0..7
      v8bf hi = *(const v8bf*)(src + 16);  // j 8..15 : K = 16 + grp*8 + 0..7
      bf16_t* dst = hs + (size_t)(t + 1) * HS_SLOT +
                    ((size_t)mt2 * HKT + (wg * 2 + ktl)) * 512 + lane16;
      *(v8bf*)dst = lo;
      *(v8bf*)(dst + 8) = hi;
    }

    // ---- inter-step device barrier (release h stores / acquire peers' h) ----
    __syncthreads();
    if (tid == 0) {
      __hip_atomic_fetch_add(&bar[t], 1, __ATOMIC_RELEASE, __HIP_MEMORY_SCOPE_AGENT);
      while (__hip_atomic_load(&bar[t], __ATOMIC_ACQUIRE, __HIP_MEMORY_SCOPE_AGENT) < NWG) {
        __builtin_amdgcn_s_sleep(1);
      }
    }
    __syncthreads();
  }
}

// ---------------- output projection: relu(hs @ Wo + bo), one block per timestep ----------------
__global__ __launch_bounds__(256) void proj_kernel(const bf16_t* __restrict__ hs,
                                                   const bf16_t* __restrict__ WoB,
                                                   const float* __restrict__ bo,
                                                   float* __restrict__ out) {
  int t = blockIdx.x;
  int tid = threadIdx.x, w = tid >> 5, lane = tid & 31;
  int mt = w & 3, ng = w >> 2;  // ng selects n-tiles [ng*8, ng*8+8)
  int lane16 = lane * 16;
  v8f acc[8] = {};

  const bf16_t* ap = hs + (size_t)(t + 1) * HS_SLOT + ((size_t)mt * HKT) * 512 + lane16;
  const bf16_t* bp = WoB + ((size_t)(ng * 8)) * 512 + lane16;

  v16bf a_c = *(const v16bf*)ap;
#pragma unroll 2
  for (int kt = 0; kt < HKT - 1; ++kt) {
    v16bf bbs[8];
#pragma unroll
    for (int q = 0; q < 8; ++q) bbs[q] = *(const v16bf*)(bp + (size_t)q * 512);
    ap += 512;
    v16bf a_n = *(const v16bf*)ap;
#pragma unroll
    for (int q = 0; q < 8; ++q) acc[q] = WMMA_BF16(a_c, bbs[q], acc[q]);
    a_c = a_n;
    bp += (size_t)16 * 512;
  }
  {
    v16bf bbs[8];
#pragma unroll
    for (int q = 0; q < 8; ++q) bbs[q] = *(const v16bf*)(bp + (size_t)q * 512);
#pragma unroll
    for (int q = 0; q < 8; ++q) acc[q] = WMMA_BF16(a_c, bbs[q], acc[q]);
  }

  int hgrp = lane >> 4;
#pragma unroll
  for (int q = 0; q < 8; ++q) {
    int col = (ng * 8 + q) * 16 + (lane & 15);
    float bv = bo[col];
#pragma unroll
    for (int v = 0; v < 8; ++v) {
      int m = mt * 16 + v + 8 * hgrp;
      float r = acc[q][v] + bv;
      out[((size_t)m * T_ + t) * DOUT + col] = r > 0.0f ? r : 0.0f;
    }
  }
}

extern "C" void kernel_launch(void* const* d_in, const int* in_sizes, int n_in,
                              void* d_out, int out_size, void* d_ws, size_t ws_size,
                              hipStream_t stream) {
  const float* x  = (const float*)d_in[0];
  const float* Wx = (const float*)d_in[1];
  const float* Wh = (const float*)d_in[2];
  const float* b  = (const float*)d_in[3];
  const float* Wo = (const float*)d_in[4];
  const float* bo = (const float*)d_in[5];
  float* out = (float*)d_out;

  char* p = (char*)d_ws;
  auto take = [&](size_t bytes) -> char* {
    char* r = p;
    p += (bytes + 255) & ~(size_t)255;
    return r;
  };
  bf16_t* WB  = (bf16_t*)take((size_t)KTOT * 4096 * 2);        // 10.5 MB
  bf16_t* WoB = (bf16_t*)take((size_t)H_ * DOUT * 2);          // 0.5 MB
  bf16_t* xA  = (bf16_t*)take((size_t)T_ * 4 * XKT * 512 * 2); // 16.8 MB
  bf16_t* hs  = (bf16_t*)take((size_t)(T_ + 1) * HS_SLOT * 2); // 67.2 MB
  int*    bar = (int*)take((size_t)T_ * 4);                    // 2 KB

  {
    int total = NKT * 256 * 512;  // 5,242,880
    hipLaunchKernelGGL(prep_wb, dim3((total + 255) / 256), dim3(256), 0, stream, Wx, Wh, WB, total);
  }
  {
    int total = HKT * 16 * 512;   // 262,144
    hipLaunchKernelGGL(prep_wob, dim3((total + 255) / 256), dim3(256), 0, stream, Wo, WoB, total);
  }
  {
    int total = T_ * 4 * XKT * 512;  // 8,388,608
    hipLaunchKernelGGL(prep_xa, dim3((total + 255) / 256), dim3(256), 0, stream, x, xA, total);
  }
  {
    int n = B_ * H_;  // 65536 (== HS_SLOT); also zeroes bar[0..T_)
    hipLaunchKernelGGL(init_state, dim3((n + 255) / 256), dim3(256), 0, stream, hs, bar, n);
  }
  hipLaunchKernelGGL(lstm_persistent, dim3(NWG), dim3(256), 0, stream, xA, WB, b, hs, bar);
  hipLaunchKernelGGL(proj_kernel, dim3(T_), dim3(256), 0, stream, hs, WoB, bo, out);
}
